// RingDilatedAttentionV3_42614665510997
// MI455X (gfx1250) — matrix operand order repaired and checked by reading, not compiled, not run on hardware
//
#include <hip/hip_runtime.h>
#include <hip/hip_bf16.h>

typedef __attribute__((ext_vector_type(2)))  __bf16 v2bf;
typedef __attribute__((ext_vector_type(8)))  __bf16 v8bf;
typedef __attribute__((ext_vector_type(16))) __bf16 v16bf;
typedef __attribute__((ext_vector_type(8)))  float  v8f;

#define N_SEQ  4096
#define NHEADS 12
#define HDIM   64
#define QTILE  64     // query rows per block
#define KTILE  64     // key rows per iteration
#define WROWS  16     // query rows per wave
#define ROWSTRIDE (NHEADS * HDIM)   // 768 floats per sequence row
// 1/sqrt(64) * log2(e): base-2 softmax with the scale folded into Q
#define QSCALE (0.125f * 1.4426950408889634f)

// Dilated gather index per head group (derived from reference _dilated_indices)
__device__ __forceinline__ int gather_idx(int grp, int j) {
    if (grp == 0) return j;                       // s=1024, r=1: identity
    if (grp == 1) {                               // s=2048, r=2, offset=1
        int seg = j >> 11, p = j & 2047;
        return (seg << 11) + ((p < 1024) ? (1 + 2 * p) : (p - 1024));
    }
    // grp == 2: s=4096, r=4, offset=2 (single segment)
    return (j < 1024) ? (2 + 4 * j) : (j - 1024);
}

// ---- Fused DPP16 butterfly reductions: one VOP2+DPP per step ----
// xor1 = quad_perm[1,0,3,2]; xor2 = quad_perm[2,3,0,1];
// xor4 ~ row_half_mirror (value quad-uniform by then);
// xor8 ~ row_mirror (value 8-lane-uniform by then).
__device__ __forceinline__ float red_max16(float x) {
    float t;
    asm("v_max_num_f32_dpp %0, %1, %2 quad_perm:[1,0,3,2] row_mask:0xf bank_mask:0xf bound_ctrl:1"
        : "=v"(t) : "v"(x), "v"(x));
    asm("v_max_num_f32_dpp %0, %1, %2 quad_perm:[2,3,0,1] row_mask:0xf bank_mask:0xf bound_ctrl:1"
        : "=v"(x) : "v"(t), "v"(t));
    asm("v_max_num_f32_dpp %0, %1, %2 row_half_mirror row_mask:0xf bank_mask:0xf bound_ctrl:1"
        : "=v"(t) : "v"(x), "v"(x));
    asm("v_max_num_f32_dpp %0, %1, %2 row_mirror row_mask:0xf bank_mask:0xf bound_ctrl:1"
        : "=v"(x) : "v"(t), "v"(t));
    return x;
}
__device__ __forceinline__ float red_sum16(float x) {
    float t;
    asm("v_add_f32_dpp %0, %1, %2 quad_perm:[1,0,3,2] row_mask:0xf bank_mask:0xf bound_ctrl:1"
        : "=v"(t) : "v"(x), "v"(x));
    asm("v_add_f32_dpp %0, %1, %2 quad_perm:[2,3,0,1] row_mask:0xf bank_mask:0xf bound_ctrl:1"
        : "=v"(x) : "v"(t), "v"(t));
    asm("v_add_f32_dpp %0, %1, %2 row_half_mirror row_mask:0xf bank_mask:0xf bound_ctrl:1"
        : "=v"(t) : "v"(x), "v"(x));
    asm("v_add_f32_dpp %0, %1, %2 row_mirror row_mask:0xf bank_mask:0xf bound_ctrl:1"
        : "=v"(x) : "v"(t), "v"(t));
    return x;
}

// Convert 8 floats (two float4) to packed bf16x8 with optional scale.
__device__ __forceinline__ v8bf pack8(float4 a, float4 b, float scale) {
    v8bf r;
    r[0] = (__bf16)(a.x * scale); r[1] = (__bf16)(a.y * scale);
    r[2] = (__bf16)(a.z * scale); r[3] = (__bf16)(a.w * scale);
    r[4] = (__bf16)(b.x * scale); r[5] = (__bf16)(b.y * scale);
    r[6] = (__bf16)(b.z * scale); r[7] = (__bf16)(b.w * scale);
    return r;
}

// Build a WMMA operand from two contiguous 16-byte LDS chunks.
__device__ __forceinline__ v16bf ldfrag(const __bf16* p0, const __bf16* p1) {
    v16bf r;
    *(v8bf*)&r       = *(const v8bf*)p0;   // elements 0..7
    *((v8bf*)&r + 1) = *(const v8bf*)p1;   // elements 8..15
    return r;
}

__global__ __launch_bounds__(128)
void dilated_flash_attn_bf16_wmma(const float* __restrict__ q,
                                  const float* __restrict__ k,
                                  const float* __restrict__ v,
                                  float* __restrict__ out) {
    __shared__ __attribute__((aligned(16))) __bf16 Qs[QTILE * HDIM];      // 8 KB, row-major, pre-scaled
    __shared__ __attribute__((aligned(16))) __bf16 KsT[HDIM * KTILE];     // 8 KB, TRANSPOSED [dim][key]
    __shared__ __attribute__((aligned(16))) __bf16 Vs[KTILE * HDIM];      // 8 KB, row-major
    __shared__ __attribute__((aligned(16))) __bf16 Ps[4][WROWS * KTILE];  // 8 KB, wave-private

    const int head  = blockIdx.y;
    const int qbase = blockIdx.x * QTILE;
    const int grp   = head >> 2;
    const int tid   = threadIdx.x;
    const int wave  = tid >> 5;
    const int lane  = tid & 31;
    const int hgrp  = lane >> 4;      // which 16-lane half
    const int mrow  = lane & 15;      // A-matrix row / C-matrix column index

    // ---- stage Q tile (fp32 -> bf16), fold in scale * log2(e) ----
    {
        int r  = tid >> 1;                 // 0..63
        int c0 = (tid & 1) * 32;
        const float4* src = (const float4*)(q + (size_t)(qbase + r) * ROWSTRIDE
                                              + head * HDIM + c0);
        #pragma unroll
        for (int j = 0; j < 4; ++j)
            *(v8bf*)&Qs[r * HDIM + c0 + 8 * j] =
                pack8(src[2 * j], src[2 * j + 1], QSCALE);
    }
    __syncthreads();

    // ---- Q A-fragments: contiguous b128 LDS loads per documented 16-bit A layout ----
    const int qrow = wave * WROWS + mrow;
    v16bf aq[2];
    #pragma unroll
    for (int s = 0; s < 2; ++s) {
        int klo = s * 32 + hgrp * 8;
        aq[s] = ldfrag(&Qs[qrow * HDIM + klo], &Qs[qrow * HDIM + klo + 16]);
    }

    float mrow_max[8], mrow_sum[8];
    v8f o[4] = {};
    #pragma unroll
    for (int r = 0; r < 8; ++r) { mrow_max[r] = -1e30f; mrow_sum[r] = 0.f; }

    for (int kt = 0; kt < N_SEQ; kt += KTILE) {
        __syncthreads();   // all waves done reading previous K/V tile
        // ---- gather + stage K (transposed) then V (row-major), vectorized ----
        {   // K: each thread converts a key-pair x 16-dim chunk
            int rp = tid >> 2;            // 0..31 -> keys 2rp, 2rp+1
            int cc = tid & 3;             // 0..3  -> dims cc*16..cc*16+15
            int r0 = 2 * rp;
            int g0 = gather_idx(grp, kt + r0);
            int g1 = gather_idx(grp, kt + r0 + 1);
            const float4* p0 = (const float4*)(k + (size_t)g0 * ROWSTRIDE + head * HDIM + cc * 16);
            const float4* p1 = (const float4*)(k + (size_t)g1 * ROWSTRIDE + head * HDIM + cc * 16);
            float ra[16], rb[16];
            #pragma unroll
            for (int j = 0; j < 4; ++j) {
                float4 t0 = p0[j], t1 = p1[j];
                ra[4 * j] = t0.x; ra[4 * j + 1] = t0.y; ra[4 * j + 2] = t0.z; ra[4 * j + 3] = t0.w;
                rb[4 * j] = t1.x; rb[4 * j + 1] = t1.y; rb[4 * j + 2] = t1.z; rb[4 * j + 3] = t1.w;
            }
            #pragma unroll
            for (int i = 0; i < 16; ++i) {
                v2bf pr; pr[0] = (__bf16)ra[i]; pr[1] = (__bf16)rb[i];
                *(v2bf*)&KsT[(cc * 16 + i) * KTILE + r0] = pr;
            }
        }
        {   // V: row-major, 16-byte LDS stores
            int r  = tid >> 1;            // 0..63
            int c0 = (tid & 1) * 32;
            int gr = gather_idx(grp, kt + r);
            const float4* src = (const float4*)(v + (size_t)gr * ROWSTRIDE + head * HDIM + c0);
            #pragma unroll
            for (int j = 0; j < 4; ++j)
                *(v8bf*)&Vs[r * HDIM + c0 + 8 * j] =
                    pack8(src[2 * j], src[2 * j + 1], 1.0f);
        }
        __syncthreads();

        // ---- S = Q * K^T : 4 N-subtiles x 2 K-slices = 8 WMMAs ----
        v8f s[4] = {};
        #pragma unroll
        for (int sl = 0; sl < 2; ++sl) {
            int kd = sl * 32 + hgrp * 8;  // B lane = key column, elements stride the K(dim) axis
            v16bf kb[4];
            #pragma unroll
            for (int n = 0; n < 4; ++n)
                #pragma unroll
                for (int e = 0; e < 16; ++e)
                    kb[n][e] = KsT[(kd + (e & 7) + ((e >> 3) << 4)) * KTILE + n * 16 + mrow];
            #pragma unroll
            for (int n = 0; n < 4; ++n)
                s[n] = __builtin_amdgcn_wmma_f32_16x16x32_bf16(false, aq[sl], false, kb[n],
                                                               (short)0, s[n], false, false);
        }

        // ---- online softmax, base-2 (each v8f slot r = one query row) ----
        float al[8];
        #pragma unroll
        for (int r = 0; r < 8; ++r) {
            float x0 = s[0][r], x1 = s[1][r], x2 = s[2][r], x3 = s[3][r];
            float tmax = red_max16(fmaxf(fmaxf(x0, x1), fmaxf(x2, x3)));
            float mn = fmaxf(mrow_max[r], tmax);
            al[r] = __builtin_exp2f(mrow_max[r] - mn);
            mrow_max[r] = mn;
            float p0 = __builtin_exp2f(x0 - mn);
            float p1 = __builtin_exp2f(x1 - mn);
            float p2 = __builtin_exp2f(x2 - mn);
            float p3 = __builtin_exp2f(x3 - mn);
            float ps = red_sum16((p0 + p1) + (p2 + p3));
            mrow_sum[r] = mrow_sum[r] * al[r] + ps;
            s[0][r] = p0; s[1][r] = p1; s[2][r] = p2; s[3][r] = p3;
        }
        #pragma unroll
        for (int t = 0; t < 4; ++t)
            #pragma unroll
            for (int r = 0; r < 8; ++r)
                o[t][r] *= al[r];

        // ---- stage P (C-layout -> row-major bf16, wave-private) ----
        #pragma unroll
        for (int r = 0; r < 8; ++r) {
            int m = r + 8 * hgrp;
            #pragma unroll
            for (int n = 0; n < 4; ++n)
                Ps[wave][m * KTILE + n * 16 + mrow] = (__bf16)s[n][r];
        }
        // ---- O += P * V : 2 K-slices x 4 N-tiles = 8 WMMAs ----
        #pragma unroll
        for (int sl = 0; sl < 2; ++sl) {
            int klo = sl * 32 + hgrp * 8;
            v16bf ap = ldfrag(&Ps[wave][mrow * KTILE + klo],
                              &Ps[wave][mrow * KTILE + klo + 16]);
            int kv = sl * 32 + hgrp * 8;  // B lane = dim column, elements stride the key axis
            v16bf bv[4];
            #pragma unroll
            for (int t = 0; t < 4; ++t)
                #pragma unroll
                for (int e = 0; e < 16; ++e)
                    bv[t][e] = Vs[(kv + (e & 7) + ((e >> 3) << 4)) * HDIM + t * 16 + mrow];
            #pragma unroll
            for (int t = 0; t < 4; ++t)
                o[t] = __builtin_amdgcn_wmma_f32_16x16x32_bf16(false, ap, false, bv[t],
                                                               (short)0, o[t], false, false);
        }
    }

    // ---- normalize (rcp + mul) and write out (b, n, h, d) ----
    float rinv[8];
    #pragma unroll
    for (int r = 0; r < 8; ++r) rinv[r] = __builtin_amdgcn_rcpf(mrow_sum[r]);
    #pragma unroll
    for (int t = 0; t < 4; ++t) {
        #pragma unroll
        for (int r = 0; r < 8; ++r) {
            int m    = r + 8 * hgrp;
            int grow = qbase + wave * WROWS + m;
            out[(size_t)grow * ROWSTRIDE + head * HDIM + t * 16 + mrow] =
                o[t][r] * rinv[r];
        }
    }
}

extern "C" void kernel_launch(void* const* d_in, const int* in_sizes, int n_in,
                              void* d_out, int out_size, void* d_ws, size_t ws_size,
                              hipStream_t stream) {
    (void)in_sizes; (void)n_in; (void)out_size; (void)d_ws; (void)ws_size;
    const float* q = (const float*)d_in[0];
    const float* k = (const float*)d_in[1];
    const float* v = (const float*)d_in[2];
    float* out = (float*)d_out;
    dim3 grid(N_SEQ / QTILE, NHEADS);
    dim3 block(128);
    hipLaunchKernelGGL(dilated_flash_attn_bf16_wmma, grid, block, 0, stream,
                       q, k, v, out);
}
